// SimplerGraphTransformer_78434692759834
// MI455X (gfx1250) — matrix operand order repaired and checked by reading, not compiled, not run on hardware
//
#include <hip/hip_runtime.h>
#include <hip/hip_bf16.h>
#include <math.h>

// ---------------------------------------------------------------------------
// SimplerGraphTransformer forward, fused for MI455X (gfx1250, wave32, WMMA).
// Shapes: bs=2, n=512, DX=128, DE=32, DY=64, NH=8, DF=16.
// ---------------------------------------------------------------------------

#define BS   2
#define NN   512
#define DXc  128
#define DEc  32
#define DYc  64

typedef __attribute__((ext_vector_type(16))) _Float16 v16h;
typedef __attribute__((ext_vector_type(8)))  _Float16 v8h;
typedef __attribute__((ext_vector_type(8)))  float    v8f;
typedef __attribute__((ext_vector_type(4)))  unsigned u32x4;
typedef __attribute__((ext_vector_type(8)))  int      i32x8;
typedef __attribute__((ext_vector_type(4)))  int      i32x4;

__device__ __forceinline__ v8f wmma_f16(v16h a, v16h b, v8f c) {
  // D = A(16x32 f16) * B(32x16 f16) + C(16x16 f32)
  return __builtin_amdgcn_wmma_f32_16x16x32_f16(
      /*neg_a=*/false, a, /*neg_b=*/false, b,
      /*c_mod=*/(short)0, c, /*reuse_a=*/false, /*reuse_b=*/false);
}

// A-matrix (16x32, 16-bit) lane layout per ISA 7.12.2:
// lane L (row = L&15, half = L>>4): element e -> k = (e&7) + ((e>>3)<<4) + (half<<3)
// => two contiguous 8-element runs at [half*8] and [16+half*8].
__device__ __forceinline__ v16h load_a_frag_f32(const float* rowk0, int half) {
  const float4* p0 = (const float4*)(rowk0 + (half << 3));
  const float4* p1 = (const float4*)(rowk0 + 16 + (half << 3));
  float4 x0 = p0[0], x1 = p0[1], x2 = p1[0], x3 = p1[1];
  v16h a;
  a[0]=(_Float16)x0.x; a[1]=(_Float16)x0.y; a[2]=(_Float16)x0.z; a[3]=(_Float16)x0.w;
  a[4]=(_Float16)x1.x; a[5]=(_Float16)x1.y; a[6]=(_Float16)x1.z; a[7]=(_Float16)x1.w;
  a[8]=(_Float16)x2.x; a[9]=(_Float16)x2.y; a[10]=(_Float16)x2.z; a[11]=(_Float16)x2.w;
  a[12]=(_Float16)x3.x; a[13]=(_Float16)x3.y; a[14]=(_Float16)x3.z; a[15]=(_Float16)x3.w;
  return a;
}

// ---------------------------------------------------------------------------
// Pre-pack a weight matrix W(K,N) f32 into WMMA B-fragments, f16, fragment-
// lane major: out[((kt*NT+nt)*32 + lane)*16 + e]. One thread per (frag,lane);
// afterwards every B fragment is one contiguous 32-byte load per lane.
// B 32x16 layout: lane half h, element e -> K = (e&7) + ((e>>3)<<4) + 8h.
// ---------------------------------------------------------------------------
__global__ __launch_bounds__(256) void pack_b_kernel(
    const float* __restrict__ W, int K, int N, _Float16* __restrict__ out) {
  int t = blockIdx.x * 256 + threadIdx.x;
  int lane = t & 31, frag = t >> 5;
  int NT = N >> 4;
  if (frag >= (K >> 5) * NT) return;
  int kt = frag / NT, nt = frag - kt * NT;
  int half = lane >> 4, ln = lane & 15;
  v16h vv;
#pragma unroll
  for (int e = 0; e < 16; ++e)
    vv[e] = (_Float16)W[(size_t)(kt * 32 + (e & 7) + ((e >> 3) << 4) + (half << 3)) * N + nt * 16 + ln];
  ((v16h*)out)[frag * 32 + lane] = vv;
}

// ---------------------------------------------------------------------------
// 128->128 WMMA GEMM with row-mask scaling, packed-f16 weights:
//   Out[r,n] = (sum_k In[r,k]*W[k,n] + bias[n]) * (rowmask[r] ? 1 : 0)
// Block = 128 threads = 4 waves, each wave owns a 16-row tile. Grid = M/64.
// ---------------------------------------------------------------------------
__global__ __launch_bounds__(128) void gemm128_kernel(
    const float* __restrict__ In, const _Float16* __restrict__ Wpk,
    const float* __restrict__ bias, const int* __restrict__ rowmask,
    float* __restrict__ Out) {
  int wave = threadIdx.x >> 5, lane = threadIdx.x & 31;
  int half = lane >> 4, ln = lane & 15;
  int row0 = blockIdx.x * 64 + wave * 16;

  const float* rp = In + (size_t)(row0 + ln) * DXc;
  v16h af[4];
#pragma unroll
  for (int kk = 0; kk < 4; ++kk) af[kk] = load_a_frag_f32(rp + kk * 32, half);

  float msk[8];
#pragma unroll
  for (int v = 0; v < 8; ++v)
    msk[v] = rowmask[row0 + v + (half << 3)] ? 1.f : 0.f;

  const v16h* WP = (const v16h*)Wpk;
  for (int nt = 0; nt < 8; ++nt) {
    int col = nt * 16 + ln;
    v8f c = {};
#pragma unroll
    for (int kk = 0; kk < 4; ++kk)
      c = wmma_f16(af[kk], WP[(kk * 8 + nt) * 32 + lane], c);
    float bb = bias[col];
#pragma unroll
    for (int v = 0; v < 8; ++v)
      Out[(size_t)(row0 + v + (half << 3)) * DXc + col] = (c[v] + bb) * msk[v];
  }
}

// ---------------------------------------------------------------------------
// Fused E pipeline per (b, 16-q-tile, 16-k-tile) block (256 thr = 8 waves):
//   E1 = (E@Wadd+b)*emask ; E2 = (E@Wmul+b)*emask  (WMMA, K=DE=32 in 1 step)
//   A  = E1 + (E2+1) * Q*K/4                        (Q,K pre-masked)
//   logits[b,q,k,h] = sum_d A                      -> ws (softmax later)
//   pre = ye1 + (ye2+1)*A  -> LDS chunk -> newE = (pre@Wout+b)*emask (WMMA)
// Q/K tiles are staged via the Tensor Data Mover (padded rows of 132 f32).
// Each wave owns 2 M-tiles (one q row x 16 k each); no cross-wave LDS deps.
// ---------------------------------------------------------------------------
__global__ __launch_bounds__(256) void fused_e_kernel(
    const float* __restrict__ E, const float* __restrict__ Q,
    const float* __restrict__ K,
    const _Float16* __restrict__ WaddPk, const float* __restrict__ badd,
    const _Float16* __restrict__ WmulPk, const float* __restrict__ bmul,
    const _Float16* __restrict__ WoutPk, const float* __restrict__ bout,
    const float* __restrict__ ye1p, const float* __restrict__ ye2p,
    const int* __restrict__ mq, const int* __restrict__ mk,
    float* __restrict__ newE, float* __restrict__ logits) {
  int b = blockIdx.z, qt = blockIdx.y, kt = blockIdx.x;
  int q0 = qt * 16, k0 = kt * 16;
  int tid = threadIdx.x, wave = tid >> 5, lane = tid & 31;
  int half = lane >> 4, ln = lane & 15;

  __shared__ __align__(32) _Float16 sPkAdd[DEc * DXc];     // 8 KB packed frags
  __shared__ __align__(32) _Float16 sPkMul[DEc * DXc];     // 8 KB
  __shared__ __align__(32) _Float16 sPkOut[DXc * DEc];     // 8 KB
  __shared__ __align__(16) float sQ[16 * 132];             // 8.25 KB (padded)
  __shared__ __align__(16) float sK[16 * 132];             // 8.25 KB
  __shared__ float sYe1[DXc], sYe2[DXc], sBa[DXc], sBm[DXc];
  __shared__ float sBo[DEc], sMq[16], sMk[16];
  __shared__ __align__(32) _Float16 sChunk[8][16 * 40];    // per-wave relayout

  // --- TDM: stage Q and K 16x128 f32 tiles into LDS with row padding -------
#if __has_builtin(__builtin_amdgcn_tensor_load_to_lds)
  if (wave == 0) {
    unsigned ldsQ = (unsigned)(size_t)(void*)&sQ[0];
    unsigned ldsK = (unsigned)(size_t)(void*)&sK[0];
    unsigned long long gaQ =
        (unsigned long long)(size_t)(Q + (size_t)(b * NN + q0) * DXc);
    unsigned long long gaK =
        (unsigned long long)(size_t)(K + (size_t)(b * NN + k0) * DXc);
    // D# group0: count=1 | lds_addr | global_addr | type=2
    u32x4 g0q = {1u, ldsQ, (unsigned)gaQ,
                 (unsigned)((gaQ >> 32) & 0x01FFFFFFull) | (2u << 30)};
    u32x4 g0k = {1u, ldsK, (unsigned)gaK,
                 (unsigned)((gaK >> 32) & 0x01FFFFFFull) | (2u << 30)};
    // D# group1: data_size=4B, pad_enable, pad every 128 dwords by 4 dwords;
    // tensor_dim0=128, tensor_dim1=16, tile_dim0=128, tile_dim1=16,
    // tensor_dim0_stride=128.
    i32x8 g1 = {(int)((2u << 16) | (1u << 20) | (6u << 22) | (3u << 25)),
                (int)(128u << 16), (int)(16u << 16), (int)(128u << 16),
                16, 128, 0, 0};
    i32x4 z4 = {0, 0, 0, 0};
#if defined(__clang_major__) && (__clang_major__ >= 23)
    i32x8 z8 = {0, 0, 0, 0, 0, 0, 0, 0};
    __builtin_amdgcn_tensor_load_to_lds(g0q, g1, z4, z4, z8, 0);
    __builtin_amdgcn_tensor_load_to_lds(g0k, g1, z4, z4, z8, 0);
#else
    __builtin_amdgcn_tensor_load_to_lds(g0q, g1, z4, z4, 0);
    __builtin_amdgcn_tensor_load_to_lds(g0k, g1, z4, z4, 0);
#endif
    __builtin_amdgcn_s_wait_tensorcnt(0);
  }
#else
  for (int i = tid; i < 16 * DXc; i += 256) {
    int r = i >> 7, c = i & 127;
    sQ[r * 132 + c] = Q[(size_t)(b * NN + q0 + r) * DXc + c];
    sK[r * 132 + c] = K[(size_t)(b * NN + k0 + r) * DXc + c];
  }
#endif

  // --- stage packed weight fragments + small vectors -----------------------
  {
    const uint4* ga = (const uint4*)WaddPk;
    const uint4* gm = (const uint4*)WmulPk;
    const uint4* go = (const uint4*)WoutPk;
    uint4* la = (uint4*)sPkAdd;
    uint4* lm = (uint4*)sPkMul;
    uint4* lo = (uint4*)sPkOut;
    for (int i = tid; i < 512; i += 256) { la[i] = ga[i]; lm[i] = gm[i]; lo[i] = go[i]; }
  }
  if (tid < DXc) {
    sYe1[tid] = ye1p[b * DXc + tid];
    sYe2[tid] = ye2p[b * DXc + tid];
    sBa[tid] = badd[tid];
    sBm[tid] = bmul[tid];
  }
  if (tid < DEc) sBo[tid] = bout[tid];
  if (tid < 16) {
    sMq[tid] = mq[b * NN + q0 + tid] ? 1.f : 0.f;
    sMk[tid] = mk[b * NN + k0 + tid] ? 1.f : 0.f;
  }
  __syncthreads();

  _Float16* chunk = &sChunk[wave][0];
  const v16h* pkA = (const v16h*)sPkAdd;
  const v16h* pkM = (const v16h*)sPkMul;
  const v16h* pkO = (const v16h*)sPkOut;

  for (int mi = 0; mi < 2; ++mi) {
    int mt = wave * 2 + mi;            // M-tile == q_local row
    int q = q0 + mt;
    float mqv = sMq[mt];

    // A fragment straight from E: pair p = mt*16 + ln -> (q_local=mt, k_local=ln)
    const float* erow = E + ((size_t)(b * NN + q) * NN + (k0 + ln)) * DEc;
    v16h ea = load_a_frag_f32(erow, half);

    v8f acc0 = {}, acc1 = {};          // e_out accumulators (j=0..15, 16..31)

    for (int kk = 0; kk < 4; ++kk) {   // 32-wide channel chunk = 2 head tiles
#pragma unroll
      for (int t2 = 0; t2 < 2; ++t2) {
        int nt = kk * 2 + t2;          // head index
        int c = nt * 16 + ln;          // channel
        v8f c1 = {}; c1 = wmma_f16(ea, pkA[nt * 32 + lane], c1);
        v8f c2 = {}; c2 = wmma_f16(ea, pkM[nt * 32 + lane], c2);

        float qv = sQ[mt * 132 + c] * 0.25f;   // 1/sqrt(DF)
        float y1 = sYe1[c], y2p1 = sYe2[c] + 1.f;
        float b1 = sBa[c], b2 = sBm[c];
        float av[8];
#pragma unroll
        for (int v = 0; v < 8; ++v) {
          int kl = v + (half << 3);            // k_local of this C element
          float em = mqv * sMk[kl];
          float e1 = (c1[v] + b1) * em;
          float e2 = (c2[v] + b2) * em;
          float qk = qv * sK[kl * 132 + c];    // Q,K already node-masked
          float a = e1 + (e2 + 1.f) * qk;
          av[v] = a;
          chunk[kl * 40 + (t2 * 16 + ln)] = (_Float16)(y1 + y2p1 * a);
        }
        // attn logits: per-head sum over the 16 channels of this tile
#pragma unroll
        for (int v = 0; v < 8; ++v) {
          float s = av[v];
          s += __shfl_xor(s, 1, 32);
          s += __shfl_xor(s, 2, 32);
          s += __shfl_xor(s, 4, 32);
          s += __shfl_xor(s, 8, 32);
          if (ln == 0) {
            int kl = v + (half << 3);
            logits[((size_t)(b * NN + q) * NN + (k0 + kl)) * 8 + nt] = s;
          }
        }
      }
      // e_out partial over this 32-channel chunk (same-wave LDS, DS in-order)
      __builtin_amdgcn_wave_barrier();
      v8h lo = *(const v8h*)&chunk[ln * 40 + (half << 3)];
      v8h hi = *(const v8h*)&chunk[ln * 40 + 16 + (half << 3)];
      v16h af = __builtin_shufflevector(lo, hi, 0, 1, 2, 3, 4, 5, 6, 7,
                                        8, 9, 10, 11, 12, 13, 14, 15);
      acc0 = wmma_f16(af, pkO[(kk * 2 + 0) * 32 + lane], acc0);
      acc1 = wmma_f16(af, pkO[(kk * 2 + 1) * 32 + lane], acc1);
      __builtin_amdgcn_wave_barrier();
    }
    // write newE tile (masked)
#pragma unroll
    for (int v = 0; v < 8; ++v) {
      int kl = v + (half << 3);
      float em = mqv * sMk[kl];
      size_t base = ((size_t)(b * NN + q) * NN + (k0 + kl)) * DEc;
      newE[base + ln]      = (acc0[v] + sBo[ln]) * em;
      newE[base + 16 + ln] = (acc1[v] + sBo[16 + ln]) * em;
    }
  }
}

// ---------------------------------------------------------------------------
// Per (b,q) row: masked softmax over k (per head), attn *= e_mask, then
// weighted_V and the y-gating -> xpre (input of x_out GEMM).
// ---------------------------------------------------------------------------
__global__ __launch_bounds__(128) void softmax_wv_kernel(
    const float* __restrict__ logits, const float* __restrict__ V,
    const int* __restrict__ mq, const int* __restrict__ mk,
    const float* __restrict__ yxa, const float* __restrict__ yxm,
    float* __restrict__ xpre) {
  int q = blockIdx.x, b = blockIdx.y;
  int tid = threadIdx.x;
  __shared__ float sAttn[NN * 8];   // 16 KB
  __shared__ float red[128];
  float mqv = mq[b * NN + q] ? 1.f : 0.f;
  int h = tid & 7, sub = tid >> 3;  // 16 sub-lanes per head

  float mx = -1e30f;
  for (int k = sub; k < NN; k += 16) {
    float l = logits[((size_t)(b * NN + q) * NN + k) * 8 + h];
    l = mk[b * NN + k] ? l : -1e9f;
    sAttn[k * 8 + h] = l;
    mx = fmaxf(mx, l);
  }
  red[tid] = mx;
  __syncthreads();
  for (int s = 64; s >= 8; s >>= 1) {
    if (tid < s) red[tid] = fmaxf(red[tid], red[tid + s]);
    __syncthreads();
  }
  float hmax = red[h];
  __syncthreads();

  float sum = 0.f;
  for (int k = sub; k < NN; k += 16) {
    float e = __expf(sAttn[k * 8 + h] - hmax);
    sAttn[k * 8 + h] = e;
    sum += e;
  }
  __syncthreads();
  red[tid] = sum;
  __syncthreads();
  for (int s = 64; s >= 8; s >>= 1) {
    if (tid < s) red[tid] += red[tid + s];
    __syncthreads();
  }
  float inv = 1.f / red[h];
  __syncthreads();

  for (int k = sub; k < NN; k += 16) {
    float mkv = mk[b * NN + k] ? 1.f : 0.f;
    sAttn[k * 8 + h] *= inv * mqv * mkv;   // softmax * e_mask
  }
  __syncthreads();

  // weighted_V: c = tid = h2*16 + d ; V stays hot in L2 (512 KB total)
  int c = tid, h2 = c >> 4;
  float acc = 0.f;
  for (int k = 0; k < NN; ++k)
    acc += sAttn[k * 8 + h2] * V[(size_t)(b * NN + k) * DXc + c];
  float a1 = yxa[b * DXc + c], a2 = yxm[b * DXc + c];
  xpre[(size_t)(b * NN + q) * DXc + c] = a1 + (a2 + 1.f) * acc;
}

// ---------------------------------------------------------------------------
// Masked mean/std/min/max of Xq/Xk over the node axis (deterministic).
// ---------------------------------------------------------------------------
__global__ __launch_bounds__(128) void node_stats_kernel(
    const float* __restrict__ Xq, const float* __restrict__ Xk,
    const int* __restrict__ mq, const int* __restrict__ mk,
    float* __restrict__ outq, float* __restrict__ outk) {
  int b = blockIdx.x, which = blockIdx.y;
  const float* X = which ? Xk : Xq;
  const int* m = which ? mk : mq;
  float* o = which ? outk : outq;
  int c = threadIdx.x;
  float s = 0.f, s2 = 0.f, mn = 1e30f, mx = -1e30f, cnt = 0.f;
  for (int i = 0; i < NN; ++i) {
    float v = X[(size_t)(b * NN + i) * DXc + c];
    float mm = m[b * NN + i] ? 1.f : 0.f;
    s += v * mm; s2 += v * v * mm; cnt += mm;
    if (mm > 0.f) { mn = fminf(mn, v); mx = fmaxf(mx, v); }
  }
  float den = (cnt == 0.f) ? 1.f : cnt;
  float mean = s / den;
  float sd = sqrtf(fmaxf(s2 / den - mean * mean, 0.f) + 1e-10f);
  if (mn > 9e29f) mn = 0.f;
  if (mx < -9e29f) mx = 0.f;
  o[b * 512 + c] = mean;
  o[b * 512 + 128 + c] = sd;
  o[b * 512 + 256 + c] = mn;
  o[b * 512 + 384 + c] = mx;
}

// ---------------------------------------------------------------------------
// E statistics, stage 1: per (b, 8-q-row slab) partial sum/sumsq/min/max per
// channel. Deterministic (fixed reduction order; stage 2 in finalize_y).
// ---------------------------------------------------------------------------
__global__ __launch_bounds__(256) void e_stats_kernel(
    const float* __restrict__ E, const int* __restrict__ mq,
    const int* __restrict__ mk, float* __restrict__ part) {
  int slab = blockIdx.x, b = blockIdx.y;
  int tid = threadIdx.x;
  int c = tid & 31, g = tid >> 5;    // 8 groups of pairs, channel c
  float s = 0.f, s2 = 0.f, mn = 1e30f, mx = -1e30f;
  for (int pp = g; pp < 8 * NN; pp += 8) {
    int qi = slab * 8 + (pp >> 9);
    int ki = pp & 511;
    bool on = (mq[b * NN + qi] != 0) && (mk[b * NN + ki] != 0);
    float v = E[((size_t)(b * NN + qi) * NN + ki) * DEc + c];
    if (on) { s += v; s2 += v * v; mn = fminf(mn, v); mx = fmaxf(mx, v); }
  }
  __shared__ float rs[256], rs2[256], rmn[256], rmx[256];
  rs[tid] = s; rs2[tid] = s2; rmn[tid] = mn; rmx[tid] = mx;
  __syncthreads();
  for (int st = 128; st >= 32; st >>= 1) {
    if (tid < st) {
      rs[tid] += rs[tid + st];
      rs2[tid] += rs2[tid + st];
      rmn[tid] = fminf(rmn[tid], rmn[tid + st]);
      rmx[tid] = fmaxf(rmx[tid], rmx[tid + st]);
    }
    __syncthreads();
  }
  if (tid < 32) {
    size_t o = (size_t)(b * 64 + slab) * 128;
    part[o + tid] = rs[tid];
    part[o + 32 + tid] = rs2[tid];
    part[o + 64 + tid] = rmn[tid];
    part[o + 96 + tid] = rmx[tid];
  }
}

// ---------------------------------------------------------------------------
// ye1/ye2/yx1/yx2 gating vectors (per batch, 128 each).
// ---------------------------------------------------------------------------
__global__ __launch_bounds__(128) void y_small_kernel(
    const float* __restrict__ y,
    const float* Wyea, const float* byea, const float* Wyem, const float* byem,
    const float* Wyxa, const float* byxa, const float* Wyxm, const float* byxm,
    float* yea, float* yem, float* yxa, float* yxm) {
  int b = blockIdx.x, c = threadIdx.x;
  float a1 = byea[c], a2 = byem[c], a3 = byxa[c], a4 = byxm[c];
  for (int j = 0; j < DYc; ++j) {
    float yv = y[b * DYc + j];
    a1 += yv * Wyea[j * DXc + c];
    a2 += yv * Wyem[j * DXc + c];
    a3 += yv * Wyxa[j * DXc + c];
    a4 += yv * Wyxm[j * DXc + c];
  }
  yea[b * DXc + c] = a1; yem[b * DXc + c] = a2;
  yxa[b * DXc + c] = a3; yxm[b * DXc + c] = a4;
}

// ---------------------------------------------------------------------------
// new_y head: stats finalize + y_proj/reduce_* + MLP.
// ---------------------------------------------------------------------------
__global__ __launch_bounds__(64) void finalize_y_kernel(
    const float* __restrict__ y, const int* __restrict__ mq,
    const int* __restrict__ mk, const float* __restrict__ xqs,
    const float* __restrict__ xks, const float* __restrict__ part,
    const float* Wyp, const float* byp,
    const float* Wrxq, const float* brxq,
    const float* Wrxk, const float* brxk,
    const float* Wre, const float* bre,
    const float* Wy0, const float* by0,
    const float* Wy1, const float* by1,
    float* __restrict__ out_y) {
  int b = blockIdx.x, t = threadIdx.x;  // 64 threads
  __shared__ float se[128], cat[256], h0[64], cnts[2];
  if (t == 0) {
    int cq = 0, ck = 0;
    for (int i = 0; i < NN; ++i) {
      cq += mq[b * NN + i] ? 1 : 0;
      ck += mk[b * NN + i] ? 1 : 0;
    }
    cnts[0] = (float)cq; cnts[1] = (float)ck;
  }
  __syncthreads();
  float cnt = cnts[0] * cnts[1];
  float den = (cnt == 0.f) ? 1.f : cnt;
  if (t < 32) {
    float s = 0.f, s2 = 0.f, mn = 1e30f, mx = -1e30f;
    for (int sl = 0; sl < 64; ++sl) {
      size_t o = (size_t)(b * 64 + sl) * 128;
      s += part[o + t]; s2 += part[o + 32 + t];
      mn = fminf(mn, part[o + 64 + t]);
      mx = fmaxf(mx, part[o + 96 + t]);
    }
    float mean = s / den;
    float sd = sqrtf(fmaxf(s2 / den - mean * mean, 0.f) + 1e-10f);
    if (mn > 9e29f) mn = 0.f;
    if (mx < -9e29f) mx = 0.f;
    se[t] = mean; se[32 + t] = sd; se[64 + t] = mn; se[96 + t] = mx;
  }
  __syncthreads();
  { float a = byp[t];  for (int j = 0; j < 64;  ++j) a += y[b*64+j]   * Wyp[j*64+t];  cat[t]       = a; }
  { float a = brxq[t]; for (int j = 0; j < 512; ++j) a += xqs[b*512+j]* Wrxq[j*64+t]; cat[64 + t]  = a; }
  { float a = brxk[t]; for (int j = 0; j < 512; ++j) a += xks[b*512+j]* Wrxk[j*64+t]; cat[128 + t] = a; }
  { float a = bre[t];  for (int j = 0; j < 128; ++j) a += se[j]       * Wre[j*64+t];  cat[192 + t] = a; }
  __syncthreads();
  { float a = by0[t]; for (int j = 0; j < 256; ++j) a += cat[j] * Wy0[j*64+t]; h0[t] = fmaxf(a, 0.f); }
  __syncthreads();
  { float a = by1[t]; for (int j = 0; j < 64; ++j) a += h0[j] * Wy1[j*64+t]; out_y[b*64+t] = a; }
}

// ---------------------------------------------------------------------------
// Launch
// ---------------------------------------------------------------------------
extern "C" void kernel_launch(void* const* d_in, const int* in_sizes, int n_in,
                              void* d_out, int out_size, void* d_ws, size_t ws_size,
                              hipStream_t stream) {
  const float* Xq = (const float*)d_in[0];
  const float* Xk = (const float*)d_in[1];
  const int*   mq = (const int*)d_in[2];
  const int*   mk = (const int*)d_in[3];
  const float* E  = (const float*)d_in[4];
  const float* y  = (const float*)d_in[5];
  auto P = [&](int i) { return (const float*)d_in[6 + i]; };
  // params flat order (W,b per layer):
  // 0 q_proj 2 k_proj 4 v_proj 6 e_add 8 e_mul 10 y_e_mul 12 y_e_add
  // 14 y_x_mul 16 y_x_add 18 y_proj 20 reduce_xq 22 reduce_xk 24 reduce_e
  // 26 x_out 28 e_out 30 y_out0 32 y_out1

  float* ws = (float*)d_ws;
  float* Qw   = ws;                 // 131072
  float* Kw   = ws + 131072;        // 131072
  float* Vw   = ws + 262144;        // 131072
  float* yea  = ws + 393216;        // 256
  float* yem  = ws + 393472;        // 256
  float* yxa  = ws + 393728;        // 256
  float* yxm  = ws + 393984;        // 256
  float* xqs  = ws + 394240;        // 1024
  float* xks  = ws + 395264;        // 1024
  float* epar = ws + 396288;        // 16384 (2 b x 64 slabs x 4 stats x 32 c)
  float* logi = ws + 412672;        // 4194304
  float* xpre = ws + 4606976;       // 131072
  _Float16* wpkQ  = (_Float16*)(ws + 4738048);   // 16384 f16 = 8192 fl
  _Float16* wpkK  = (_Float16*)(ws + 4746240);
  _Float16* wpkV  = (_Float16*)(ws + 4754432);
  _Float16* wpkXO = (_Float16*)(ws + 4762624);
  _Float16* wpkEA = (_Float16*)(ws + 4770816);   // 4096 f16 = 2048 fl
  _Float16* wpkEM = (_Float16*)(ws + 4772864);
  _Float16* wpkEO = (_Float16*)(ws + 4774912);   // end ~4,776,960 fl (~19.1MB)

  float* outX = (float*)d_out;                          // 2*512*128
  float* outE = outX + 131072;                          // 2*512*512*32
  float* outY = outE + 16777216;                        // 2*64

  // Pre-pack WMMA B-fragments for all WMMA-consumed weights (f16).
  pack_b_kernel<<<4, 256, 0, stream>>>(P(0), 128, 128, wpkQ);
  pack_b_kernel<<<4, 256, 0, stream>>>(P(2), 128, 128, wpkK);
  pack_b_kernel<<<4, 256, 0, stream>>>(P(4), 128, 128, wpkV);
  pack_b_kernel<<<4, 256, 0, stream>>>(P(26), 128, 128, wpkXO);
  pack_b_kernel<<<1, 256, 0, stream>>>(P(6), 32, 128, wpkEA);
  pack_b_kernel<<<1, 256, 0, stream>>>(P(8), 32, 128, wpkEM);
  pack_b_kernel<<<1, 256, 0, stream>>>(P(28), 128, 32, wpkEO);

  // Projections (node-masked): Q/K/V
  gemm128_kernel<<<16, 128, 0, stream>>>(Xq, wpkQ, P(1), mq, Qw);
  gemm128_kernel<<<16, 128, 0, stream>>>(Xk, wpkK, P(3), mk, Kw);
  gemm128_kernel<<<16, 128, 0, stream>>>(Xk, wpkV, P(5), mk, Vw);

  // Stats + y-gating vectors
  node_stats_kernel<<<dim3(2, 2), 128, 0, stream>>>(Xq, Xk, mq, mk, xqs, xks);
  e_stats_kernel<<<dim3(64, 2), 256, 0, stream>>>(E, mq, mk, epar);
  y_small_kernel<<<2, 128, 0, stream>>>(y, P(12), P(13), P(10), P(11),
                                        P(16), P(17), P(14), P(15),
                                        yea, yem, yxa, yxm);
  finalize_y_kernel<<<2, 64, 0, stream>>>(y, mq, mk, xqs, xks, epar,
                                          P(18), P(19), P(20), P(21),
                                          P(22), P(23), P(24), P(25),
                                          P(30), P(31), P(32), P(33), outY);

  // Fused E pipeline: newE + attn logits
  fused_e_kernel<<<dim3(32, 32, 2), 256, 0, stream>>>(
      E, Qw, Kw, wpkEA, P(7), wpkEM, P(9), wpkEO, P(29),
      yea, yem, mq, mk, outE, logi);

  // Softmax + weighted V + y-gating -> xpre ; then x_out projection -> newX
  softmax_wv_kernel<<<dim3(512, 2), 128, 0, stream>>>(logi, Vw, mq, mk,
                                                      yxa, yxm, xpre);
  gemm128_kernel<<<16, 128, 0, stream>>>(xpre, wpkXO, P(27), mq, outX);
}
